// Semantic_18013092839758
// MI455X (gfx1250) — matrix operand order repaired and checked by reading, not compiled, not run on hardware
//
#include <hip/hip_runtime.h>

// ---------------------------------------------------------------------------
// Problem constants (from reference): N=50000, R=5, C=256, D=128, HEADS=8
// GEMM view: M = N*R = 250000 rows, K = C = 256, per-head Ncols = D = 128.
// ---------------------------------------------------------------------------
#define NN        50000
#define RR        5
#define CC        256
#define DD        128
#define HEADS     8
#define M_TOTAL   (NN * RR)          // 250000
#define ROWS_PB   64                 // rows of H per block (4 subtiles of 16)
#define NSUB      (ROWS_PB / 16)     // 4
#define LDS_STR   264                // padded ushort stride (256 + 8) -> conflict-free
#define BPACK_ELEMS (HEADS * CC * DD) // 262144 bf16 values

typedef __attribute__((ext_vector_type(16))) __bf16        v16bf;
typedef __attribute__((ext_vector_type(8)))  float         v8f;
typedef __attribute__((ext_vector_type(4)))  unsigned int  u32x4;
typedef __attribute__((ext_vector_type(4)))  float         f32x4;
typedef __attribute__((ext_vector_type(4)))  unsigned short u16x4;

struct Frag32 { u32x4 lo; u32x4 hi; };   // 32 bytes == v16bf

__device__ __forceinline__ unsigned short f2bf(float x) {
    unsigned int u = __float_as_uint(x);
    unsigned int r = (u + 0x7FFFu + ((u >> 16) & 1u)) >> 16;   // round-to-nearest-even
    return (unsigned short)r;
}

// Branchless tanh: 1 - 2/(exp(2x)+1).  v_exp_f32 / v_rcp_f32 are TRANS ops
// that co-execute with XDL WMMA; saturates correctly at +/-inf.
__device__ __forceinline__ float fast_tanh(float x) {
    float e = __expf(2.0f * x);
    return 1.0f - 2.0f * __builtin_amdgcn_rcpf(e + 1.0f);
}

// ---------------------------------------------------------------------------
// Kernel 1: zero score accumulators; pack W1 (f32) into bf16 WMMA B-fragment
// layout. B 32x16 (KxN) bf16 fragment: lane L holds column N = L%16 and
// 16 contiguous K values: K = k0 + (L<16 ? 0 : 16) + e  (e = 0..15).
// Bpack index = (((h*8 + nt)*8 + ks)*32 + L)*16 + e
// ---------------------------------------------------------------------------
__global__ void pack_w1_kernel(const float* __restrict__ W1,
                               unsigned short* __restrict__ Bpack,
                               float* __restrict__ scoresum) {
    int idx = blockIdx.x * blockDim.x + threadIdx.x;
    if (idx < HEADS * RR) scoresum[idx] = 0.0f;
    if (idx >= BPACK_ELEMS) return;
    int e  = idx & 15;
    int L  = (idx >> 4) & 31;
    int ks = (idx >> 9) & 7;
    int nt = (idx >> 12) & 7;
    int h  = (idx >> 15) & 7;
    int col = nt * 16 + (L & 15);
    int k   = ks * 32 + ((L < 16) ? 0 : 16) + e;
    Bpack[idx] = f2bf(W1[(h * CC + k) * DD + col]);
}

// ---------------------------------------------------------------------------
// Kernel 2: fused score GEMM.  Each block: 8 waves == 8 heads, 64 rows of H.
//   s = tanh(Htile @ W1_h + b1_h); partial[r] += (s . w2_h) for the row's r.
// A bf16 16x32 fragment: lane L -> row M = L%16, kbase = (L<16 ? 0 : 16),
//   e=0..15 -> K = k0+kbase+e (same contiguous K order as Bpack)
//   => two contiguous 16B chunks at k0+kbase and k0+kbase+8.
// C/D v8f: VGPR v, lanes 0-15 -> row v, lanes 16-31 -> row v+8; col = L%16.
// ---------------------------------------------------------------------------
__global__ __launch_bounds__(256)
void attn_score_kernel(const float* __restrict__ H,
                       const unsigned short* __restrict__ Bpack,
                       const float* __restrict__ b1,
                       const float* __restrict__ w2,
                       float* __restrict__ scoresum) {
    __shared__ unsigned short As[ROWS_PB * LDS_STR];   // 33.8 KB bf16 A tile
    __shared__ float sacc[HEADS * RR];                 // per-block score partials

    const int tid  = threadIdx.x;
    const int wave = tid >> 5;            // == head
    const int lane = tid & 31;
    const long row0 = (long)blockIdx.x * ROWS_PB;

    if (tid < HEADS * RR) sacc[tid] = 0.0f;

    // --- cooperative load: 64 x 256 f32 -> bf16 in LDS (vectorized x4) ---
    for (int i = tid; i < ROWS_PB * (CC / 4); i += 256) {
        int r  = i >> 6;            // row in tile
        int c4 = i & 63;            // float4 column
        f32x4 v = {0.f, 0.f, 0.f, 0.f};
        long grow = row0 + r;
        if (grow < M_TOTAL) v = ((const f32x4*)H)[grow * (CC / 4) + c4];
        u16x4 b;
        b.x = f2bf(v.x); b.y = f2bf(v.y); b.z = f2bf(v.z); b.w = f2bf(v.w);
        *(u16x4*)&As[r * LDS_STR + c4 * 4] = b;
    }
    __syncthreads();

    const int h  = wave;
    const int kb = (lane < 16) ? 0 : 16;  // A fragment K base per lane half

    float rowacc[NSUB][8];
#pragma unroll
    for (int s = 0; s < NSUB; ++s)
#pragma unroll
        for (int v = 0; v < 8; ++v) rowacc[s][v] = 0.0f;

    for (int nt = 0; nt < 8; ++nt) {      // 8 column tiles of D=128 (dynamic loop)
        const int col = nt * 16 + (lane & 15);
        const float b1v = b1[h * DD + col];
        const float w2v = w2[h * DD + col];

        // register-cache the 8 B fragments for this (head, ntile)
        v16bf Bf[8];
        const unsigned short* bp =
            Bpack + (size_t)(((h * 8 + nt) * 8) * 32 + lane) * 16;
#pragma unroll
        for (int ks = 0; ks < 8; ++ks) {
            Frag32 f;
            f.lo = *(const u32x4*)(bp + ks * 512);
            f.hi = *(const u32x4*)(bp + ks * 512 + 8);
            Bf[ks] = __builtin_bit_cast(v16bf, f);
        }

#pragma unroll
        for (int s = 0; s < NSUB; ++s) {  // 4 row subtiles of 16
            const unsigned short* ap = &As[(s * 16 + (lane & 15)) * LDS_STR];
            // load all 8 A fragments for this subtile (overlappable ds loads)
            v16bf Af[8];
#pragma unroll
            for (int ks = 0; ks < 8; ++ks) {
                Frag32 fa;
                fa.lo = *(const u32x4*)(ap + ks * 32 + kb);
                fa.hi = *(const u32x4*)(ap + ks * 32 + kb + 8);
                Af[ks] = __builtin_bit_cast(v16bf, fa);
            }
            v8f acc = {0.f, 0.f, 0.f, 0.f, 0.f, 0.f, 0.f, 0.f};
#pragma unroll
            for (int ks = 0; ks < 8; ++ks) {
                acc = __builtin_amdgcn_wmma_f32_16x16x32_bf16(
                          false, Af[ks], false, Bf[ks], (short)0, acc, false, false);
            }
            // fused epilogue: branchless tanh + dot(w2) partials per row
#pragma unroll
            for (int v = 0; v < 8; ++v) {
                float t = fast_tanh(acc[v] + b1v);
                rowacc[s][v] += t * w2v;
            }
        }
    }

    // --- reduce each 16-lane half (lanes 0-15 = row v, 16-31 = row v+8) ---
#pragma unroll
    for (int s = 0; s < NSUB; ++s) {
#pragma unroll
        for (int v = 0; v < 8; ++v) {
            float val = rowacc[s][v];
            val += __shfl_xor(val, 1);
            val += __shfl_xor(val, 2);
            val += __shfl_xor(val, 4);
            val += __shfl_xor(val, 8);
            if (lane == 0 || lane == 16) {
                long grow = row0 + s * 16 + v + ((lane == 16) ? 8 : 0);
                if (grow < M_TOTAL) {
                    int r = (int)(grow % RR);
                    atomicAdd(&sacc[h * RR + r], val);
                }
            }
        }
    }
    __syncthreads();
    if (tid < HEADS * RR) atomicAdd(&scoresum[tid], sacc[tid]);
}

// ---------------------------------------------------------------------------
// Kernel 3: scores -> mean over N -> softmax over R per head -> gamma[r]
// ---------------------------------------------------------------------------
__global__ void gamma_kernel(const float* __restrict__ scoresum,
                             float* __restrict__ gamma) {
    if (threadIdx.x != 0 || blockIdx.x != 0) return;
    float g[RR] = {0.f, 0.f, 0.f, 0.f, 0.f};
    for (int h = 0; h < HEADS; ++h) {
        float sc[RR];
        float mx = -3.4e38f;
        for (int r = 0; r < RR; ++r) {
            sc[r] = scoresum[h * RR + r] * (1.0f / (float)NN);
            mx = fmaxf(mx, sc[r]);
        }
        float se = 0.f;
        for (int r = 0; r < RR; ++r) { sc[r] = __expf(sc[r] - mx); se += sc[r]; }
        float inv = (1.0f / (float)HEADS) / se;
        for (int r = 0; r < RR; ++r) g[r] += sc[r] * inv;
    }
    for (int r = 0; r < RR; ++r) gamma[r] = g[r];
}

// ---------------------------------------------------------------------------
// Kernel 4: out[n,c] = sum_r gamma[r] * H[n,r,c]   (memory-bound, float4)
// ---------------------------------------------------------------------------
__global__ void combine_kernel(const float* __restrict__ H,
                               const float* __restrict__ gamma,
                               float* __restrict__ out) {
    int idx = blockIdx.x * blockDim.x + threadIdx.x;
    if (idx >= NN * (CC / 4)) return;
    long n  = idx >> 6;           // CC/4 = 64
    int  c4 = idx & 63;
    const f32x4* Hb = (const f32x4*)H + n * (RR * (CC / 4)) + c4;
    f32x4 a = {0.f, 0.f, 0.f, 0.f};
#pragma unroll
    for (int r = 0; r < RR; ++r) {
        float g  = gamma[r];
        f32x4 hv = Hb[r * (CC / 4)];
        a.x += g * hv.x; a.y += g * hv.y; a.z += g * hv.z; a.w += g * hv.w;
    }
    ((f32x4*)out)[idx] = a;
}

// ---------------------------------------------------------------------------
// Launch: pack -> score GEMM -> gamma -> combine (all on `stream`)
// ws layout: [0,160)   scoresum (40 f32)
//            [256,276) gamma    (5 f32)
//            [512, 512+512KB) Bpack (bf16)
// ---------------------------------------------------------------------------
extern "C" void kernel_launch(void* const* d_in, const int* in_sizes, int n_in,
                              void* d_out, int out_size, void* d_ws, size_t ws_size,
                              hipStream_t stream) {
    const float* H  = (const float*)d_in[0];
    const float* W1 = (const float*)d_in[1];
    const float* b1 = (const float*)d_in[2];
    const float* w2 = (const float*)d_in[3];
    float* out = (float*)d_out;

    float* scoresum       = (float*)d_ws;
    float* gamma          = (float*)((char*)d_ws + 256);
    unsigned short* Bpack = (unsigned short*)((char*)d_ws + 512);

    (void)in_sizes; (void)n_in; (void)out_size; (void)ws_size;

    // 1) pack W1 to bf16 fragments + zero score accumulators
    pack_w1_kernel<<<BPACK_ELEMS / 256, 256, 0, stream>>>(W1, Bpack, scoresum);

    // 2) fused score GEMM (WMMA bf16)
    int blocks = (M_TOTAL + ROWS_PB - 1) / ROWS_PB;   // 3907
    attn_score_kernel<<<blocks, 256, 0, stream>>>(H, Bpack, b1, w2, scoresum);

    // 3) softmax -> gamma
    gamma_kernel<<<1, 32, 0, stream>>>(scoresum, gamma);

    // 4) weighted combine (fully writes d_out)
    int cblocks = (NN * (CC / 4) + 255) / 256;        // 12500
    combine_kernel<<<cblocks, 256, 0, stream>>>(H, gamma, out);
}
